// ComplexValuedEngine_12395275616274
// MI455X (gfx1250) — matrix operand order repaired and checked by reading, not compiled, not run on hardware
//
#include <hip/hip_runtime.h>
#include <cmath>

// ---------------- problem constants ----------------
#define N_CELLS 2048
#define D_IN    512
#define D_HID   1024
#define D_OUT   512
#define EIN     1536            // D_IN + D_HID
#define GIN     1537            // D_OUT + 1 + D_HID
#define GIN_PAD 1568            // 49 * 32, zero-padded K for WMMA
#define NFAC    8
#define FS      256             // N_CELLS / NFAC
#define DC      64              // FS / 4
#define SYNC_W   0.15f
#define DEBATE_W 0.15f

typedef _Float16 half16 __attribute__((ext_vector_type(16)));
typedef _Float16 half8  __attribute__((ext_vector_type(8)));
typedef float    float8 __attribute__((ext_vector_type(8)));

#define GS_LOOP(i, n) for (size_t i = blockIdx.x * (size_t)blockDim.x + threadIdx.x; \
                           i < (size_t)(n); i += (size_t)gridDim.x * blockDim.x)

// ---------------- WMMA fragment helpers (wave32, 16x16x32 f16) ----------------
// A (16x32, row-major source): lanes 0-15 rows M=0..15 hold K {0..15},
// lanes 16-31 same rows hold K {16..31}
__device__ __forceinline__ half16 load_a_frag(const _Float16* A, int ld, int row,
                                              int k0, int hi) {
  const _Float16* base = A + (size_t)row * ld + k0 + hi * 16;
  half8 lo = *(const half8*)(base);
  half8 hh = *(const half8*)(base + 8);
  half16 r;
#pragma unroll
  for (int i = 0; i < 8; ++i) { r[i] = lo[i]; r[i + 8] = hh[i]; }
  return r;
}
// B (32x16): weights pre-transposed to [N x K] so a lane's fragment
// (col = lane%16, K = k0 + (lane/16)*16 .. +15) is one contiguous 32B load.
__device__ __forceinline__ half16 load_b_frag(const _Float16* Bt, int ld, int col,
                                              int k0, int hi) {
  return *(const half16*)(Bt + (size_t)col * ld + k0 + hi * 16);
}
__device__ __forceinline__ float8 wmma_f16(half16 a, half16 b, float8 c) {
  return __builtin_amdgcn_wmma_f32_16x16x32_f16(false, a, false, b, (short)0, c,
                                                false, false);
}

// ---------------- conversion / prep kernels ----------------
__global__ void k_conv_h(const float* h_re, const float* h_im,
                         _Float16* hre16, _Float16* him16) {
  GS_LOOP(i, (size_t)N_CELLS * D_HID) {
    hre16[i] = (_Float16)h_re[i];
    him16[i] = (_Float16)h_im[i];
  }
}

// dW = ea_w - eg_w restricted to the h-rows [512..1535], transposed to [N x K] f16
__global__ void k_conv_engine_w(const float* ea_wr, const float* eg_wr,
                                const float* ea_wi, const float* eg_wi,
                                _Float16* wrT, _Float16* wiT) {
  GS_LOOP(i, (size_t)D_HID * D_OUT) {
    int n = (int)(i % D_OUT);
    int k = (int)(i / D_OUT);
    size_t src = (size_t)(D_IN + k) * D_OUT + n;
    wrT[(size_t)n * D_HID + k] = (_Float16)(ea_wr[src] - eg_wr[src]);
    wiT[(size_t)n * D_HID + k] = (_Float16)(ea_wi[src] - eg_wi[src]);
  }
}

// generic [K_src x N] f32 -> [N x K_pad] f16 transpose with zero pad
__global__ void k_conv_wT_pad(const float* w, _Float16* wT, int K_src, int N, int K_pad) {
  GS_LOOP(i, (size_t)N * K_pad) {
    int n = (int)(i % N);
    int k = (int)(i / N);
    _Float16 v = (_Float16)0.f;
    if (k < K_src) v = (_Float16)w[(size_t)k * N + n];
    wT[(size_t)n * K_pad + k] = v;
  }
}

// xa[n] = x @ dW[0:512, n] + folded complex biases
__global__ void k_xa(const float* x,
                     const float* ea_wr, const float* eg_wr,
                     const float* ea_wi, const float* eg_wi,
                     const float* ea_br, const float* eg_br,
                     const float* ea_bi, const float* eg_bi,
                     float* xa_re, float* xa_im) {
  int n = blockIdx.x * blockDim.x + threadIdx.x;
  if (n >= D_OUT) return;
  float sr = 0.f, si = 0.f;
  for (int k = 0; k < D_IN; ++k) {
    float xv = x[k];
    size_t idx = (size_t)k * D_OUT + n;
    sr += xv * (ea_wr[idx] - eg_wr[idx]);
    si += xv * (ea_wi[idx] - eg_wi[idx]);
  }
  float br = ea_br[n] - eg_br[n];
  float bi = ea_bi[n] - eg_bi[n];
  xa_re[n] = sr + br - bi;   // o_re bias: (ea_br-eg_br) - (ea_bi-eg_bi)
  xa_im[n] = si + br + bi;   // o_im bias: (ea_br-eg_br) + (ea_bi-eg_bi)
}

// ---------------- engine GEMM: o_re/o_im = complex (h) x dW ----------------
// 4 fused GEMMs [2048x1024]x[1024x512]; 2x1 register blocking (32x16 per wave)
// shares the two B weight streams across two M tiles; 3 accumulators per tile
// avoid A/B negation (f16 WMMA NEG supports C only).
__global__ void __launch_bounds__(256, 1)
k_engine_gemm(const _Float16* __restrict__ hre16,
              const _Float16* __restrict__ him16,
              const _Float16* __restrict__ wrT,
              const _Float16* __restrict__ wiT,
              const float* xa_re, const float* xa_im,
              float* o_re, float* o_im) {
  const int lane = threadIdx.x & 31;
  const int wave = threadIdx.x >> 5;
  const int tile = blockIdx.x * 8 + wave;     // 2048 wave-tiles = 64(M) x 32(N)
  const int tN = tile & 31;
  const int tM = tile >> 5;
  const int lid = lane & 15, hi = lane >> 4;
  const int row0 = tM * 32, col0 = tN * 16;
  float8 accP[2] = {};
  float8 accQ[2] = {};
  float8 accI[2] = {};
  for (int k0 = 0; k0 < D_HID; k0 += 32) {
    half16 bR = load_b_frag(wrT, D_HID, col0 + lid, k0, hi);
    half16 bI = load_b_frag(wiT, D_HID, col0 + lid, k0, hi);
#pragma unroll
    for (int mi = 0; mi < 2; ++mi) {
      half16 aRe = load_a_frag(hre16, D_HID, row0 + mi * 16 + lid, k0, hi);
      half16 aIm = load_a_frag(him16, D_HID, row0 + mi * 16 + lid, k0, hi);
      accP[mi] = wmma_f16(aRe, bR, accP[mi]);  // h_re @ dWr
      accQ[mi] = wmma_f16(aIm, bI, accQ[mi]);  // h_im @ dWi (subtract in epilogue)
      accI[mi] = wmma_f16(aIm, bR, accI[mi]);  // h_im @ dWr
      accI[mi] = wmma_f16(aRe, bI, accI[mi]);  // + h_re @ dWi
    }
  }
  const float xr = xa_re[col0 + lid];
  const float xi = xa_im[col0 + lid];
#pragma unroll
  for (int mi = 0; mi < 2; ++mi)
#pragma unroll
    for (int v = 0; v < 8; ++v) {
      int rr = row0 + mi * 16 + hi * 8 + v;    // C layout: VGPR v -> M = v + 8*(lane/16)
      size_t idx = (size_t)rr * D_OUT + col0 + lid;
      o_re[idx] = accP[mi][v] - accQ[mi][v] + xr;
      o_im[idx] = accI[mi][v] + xi;
    }
}

// tension[m] = mean_j(o_re^2 + o_im^2)
__global__ void k_tension(const float* o_re, const float* o_im, float* tension) {
  __shared__ float red[256];
  int m = blockIdx.x, t = threadIdx.x;
  float s = 0.f;
  for (int j = t; j < D_OUT; j += 256) {
    float a = o_re[(size_t)m * D_OUT + j];
    float b = o_im[(size_t)m * D_OUT + j];
    s += a * a + b * b;
  }
  red[t] = s; __syncthreads();
  for (int k = 128; k > 0; k >>= 1) { if (t < k) red[t] += red[t + k]; __syncthreads(); }
  if (t == 0) tension[m] = red[0] / (float)D_OUT;
}

// cm = [ |mem| , |h| ] in f16, zero-padded to GIN_PAD
__global__ void k_build_cm(const float* o_re, const float* o_im, const float* tension,
                           const float* h_re, const float* h_im, _Float16* cm16) {
  GS_LOOP(i, (size_t)N_CELLS * GIN_PAD) {
    int m = (int)(i / GIN_PAD);
    int j = (int)(i % GIN_PAD);
    float v = 0.f;
    if (j < D_OUT) {
      float a = o_re[(size_t)m * D_OUT + j], b = o_im[(size_t)m * D_OUT + j];
      v = sqrtf(a * a + b * b);
    } else if (j == D_OUT) {
      v = tension[m];                    // sqrt(t^2 + 0), t >= 0
    } else if (j < GIN) {
      int d = j - (D_OUT + 1);
      float a = h_re[(size_t)m * D_HID + d], b = h_im[(size_t)m * D_HID + d];
      v = sqrtf(a * a + b * b);
    }
    cm16[i] = (_Float16)v;
  }
}

// z,r gates: two GEMMs sharing A, 2x1 blocking, sigmoid epilogue
__global__ void __launch_bounds__(256, 1)
k_zr_gemm(const _Float16* __restrict__ cm16,
          const _Float16* __restrict__ gzT,
          const _Float16* __restrict__ grT,
          const float* gz_b, const float* gr_b,
          float* z, float* r) {
  const int lane = threadIdx.x & 31;
  const int wave = threadIdx.x >> 5;
  const int tile = blockIdx.x * 8 + wave;     // 4096 wave-tiles = 64(M) x 64(N)
  const int tN = tile & 63;
  const int tM = tile >> 6;
  const int lid = lane & 15, hi = lane >> 4;
  const int row0 = tM * 32, col0 = tN * 16;
  float8 accZ[2] = {};
  float8 accR[2] = {};
  for (int k0 = 0; k0 < GIN_PAD; k0 += 32) {
    half16 bz = load_b_frag(gzT, GIN_PAD, col0 + lid, k0, hi);
    half16 br = load_b_frag(grT, GIN_PAD, col0 + lid, k0, hi);
#pragma unroll
    for (int mi = 0; mi < 2; ++mi) {
      half16 a = load_a_frag(cm16, GIN_PAD, row0 + mi * 16 + lid, k0, hi);
      accZ[mi] = wmma_f16(a, bz, accZ[mi]);
      accR[mi] = wmma_f16(a, br, accR[mi]);
    }
  }
  const float bzv = gz_b[col0 + lid], brv = gr_b[col0 + lid];
#pragma unroll
  for (int mi = 0; mi < 2; ++mi)
#pragma unroll
    for (int v = 0; v < 8; ++v) {
      int rr = row0 + mi * 16 + hi * 8 + v;
      size_t idx = (size_t)rr * D_HID + col0 + lid;
      z[idx] = 1.f / (1.f + expf(-(accZ[mi][v] + bzv)));
      r[idx] = 1.f / (1.f + expf(-(accR[mi][v] + brv)));
    }
}

// comb_r = [o_re, tension, r*h_re], comb_i = [o_im, 0, r*h_im], f16 padded
__global__ void k_build_comb(const float* o_re, const float* o_im, const float* tension,
                             const float* r, const float* h_re, const float* h_im,
                             _Float16* combr16, _Float16* combi16) {
  GS_LOOP(i, (size_t)N_CELLS * GIN_PAD) {
    int m = (int)(i / GIN_PAD);
    int j = (int)(i % GIN_PAD);
    float cr = 0.f, ci = 0.f;
    if (j < D_OUT) {
      cr = o_re[(size_t)m * D_OUT + j];
      ci = o_im[(size_t)m * D_OUT + j];
    } else if (j == D_OUT) {
      cr = tension[m];
    } else if (j < GIN) {
      int d = j - (D_OUT + 1);
      float rv = r[(size_t)m * D_HID + d];
      cr = rv * h_re[(size_t)m * D_HID + d];
      ci = rv * h_im[(size_t)m * D_HID + d];
    }
    combr16[i] = (_Float16)cr;
    combi16[i] = (_Float16)ci;
  }
}

// candidate GEMMs (2x1 blocking) + GRU blend epilogue -> nh (pre-sync) into d_out
__global__ void __launch_bounds__(256, 1)
k_cand_gemm(const _Float16* __restrict__ combr16,
            const _Float16* __restrict__ combi16,
            const _Float16* __restrict__ ghrT,
            const _Float16* __restrict__ ghiT,
            const float* ghr_b, const float* ghi_b,
            const float* z, const float* h_re, const float* h_im,
            float* nh_re, float* nh_im) {
  const int lane = threadIdx.x & 31;
  const int wave = threadIdx.x >> 5;
  const int tile = blockIdx.x * 8 + wave;     // 4096 wave-tiles = 64(M) x 64(N)
  const int tN = tile & 63;
  const int tM = tile >> 6;
  const int lid = lane & 15, hi = lane >> 4;
  const int row0 = tM * 32, col0 = tN * 16;
  float8 accR[2] = {};
  float8 accI[2] = {};
  for (int k0 = 0; k0 < GIN_PAD; k0 += 32) {
    half16 br = load_b_frag(ghrT, GIN_PAD, col0 + lid, k0, hi);
    half16 bi = load_b_frag(ghiT, GIN_PAD, col0 + lid, k0, hi);
#pragma unroll
    for (int mi = 0; mi < 2; ++mi) {
      half16 ar = load_a_frag(combr16, GIN_PAD, row0 + mi * 16 + lid, k0, hi);
      half16 ai = load_a_frag(combi16, GIN_PAD, row0 + mi * 16 + lid, k0, hi);
      accR[mi] = wmma_f16(ar, br, accR[mi]);
      accI[mi] = wmma_f16(ai, bi, accI[mi]);
    }
  }
  const float bhr = ghr_b[col0 + lid], bhi = ghi_b[col0 + lid];
#pragma unroll
  for (int mi = 0; mi < 2; ++mi)
#pragma unroll
    for (int v = 0; v < 8; ++v) {
      int rr = row0 + mi * 16 + hi * 8 + v;
      size_t idx = (size_t)rr * D_HID + col0 + lid;
      float zz = z[idx];
      float cr = tanhf(accR[mi][v] + bhr);
      float ci = tanhf(accI[mi][v] + bhi);
      nh_re[idx] = (1.f - zz) * h_re[idx] + zz * cr;
      nh_im[idx] = (1.f - zz) * h_im[idx] + zz * ci;
    }
}

// faction means of pre-sync nh: fmean[(c*NFAC+f)*D_HID + col]
__global__ void k_fmean(const float* nh_re, const float* nh_im, float* fmean) {
  int f = blockIdx.x & 7;
  int c = blockIdx.x >> 3;
  const float* src = c ? nh_im : nh_re;
  for (int col = threadIdx.x; col < D_HID; col += 256) {
    float s = 0.f;
    for (int row = 0; row < FS; ++row)
      s += src[(size_t)(f * FS + row) * D_HID + col];
    fmean[(size_t)(c * NFAC + f) * D_HID + col] = s / (float)FS;
  }
}

__global__ void k_gmean(const float* fmean, float* gmean) {
  GS_LOOP(i, 2 * D_HID) {
    int c = (int)(i / D_HID), col = (int)(i % D_HID);
    float s = 0.f;
    for (int f = 0; f < NFAC; ++f) s += fmean[(size_t)(c * NFAC + f) * D_HID + col];
    gmean[i] = s / (float)NFAC;
  }
}

// in-place sync + debate blend over d_out nh regions
__global__ void k_blend(float* nh_re, float* nh_im, const float* fmean,
                        const float* gmean, const int* step) {
  const bool debate = (*step > 5);
  GS_LOOP(i, (size_t)2 * N_CELLS * D_HID) {
    int c = (int)(i / ((size_t)N_CELLS * D_HID));
    size_t rem = i % ((size_t)N_CELLS * D_HID);
    int m = (int)(rem / D_HID), d = (int)(rem % D_HID);
    int f = m / FS, rif = m % FS;
    float* dst = c ? nh_im : nh_re;
    float v = dst[rem];
    v = (1.f - SYNC_W) * v + SYNC_W * fmean[(size_t)(c * NFAC + f) * D_HID + d];
    if (debate && rif < DC)
      v = (1.f - DEBATE_W) * v + DEBATE_W * gmean[(size_t)c * D_HID + d];
    dst[rem] = v;
  }
}

// softmax over tension + mean(tension)
__global__ void k_softmax(const float* tension, float* wsoft, float* tmean_out) {
  __shared__ float red[256];
  int t = threadIdx.x;
  float mx = -1e30f;
  for (int i = t; i < N_CELLS; i += 256) mx = fmaxf(mx, tension[i]);
  red[t] = mx; __syncthreads();
  for (int k = 128; k > 0; k >>= 1) { if (t < k) red[t] = fmaxf(red[t], red[t + k]); __syncthreads(); }
  mx = red[0]; __syncthreads();
  float se = 0.f, st = 0.f;
  for (int i = t; i < N_CELLS; i += 256) { float tv = tension[i]; se += expf(tv - mx); st += tv; }
  red[t] = se; __syncthreads();
  for (int k = 128; k > 0; k >>= 1) { if (t < k) red[t] += red[t + k]; __syncthreads(); }
  se = red[0]; __syncthreads();
  red[t] = st; __syncthreads();
  for (int k = 128; k > 0; k >>= 1) { if (t < k) red[t] += red[t + k]; __syncthreads(); }
  st = red[0];
  float inv = 1.f / se;
  for (int i = t; i < N_CELLS; i += 256) wsoft[i] = expf(tension[i] - mx) * inv;
  if (t == 0) *tmean_out = st / (float)N_CELLS;
}

// out_real[j] = sum_n wsoft[n] * (j<512 ? o_re : o_im)[n, j%512]
__global__ void k_wsum(const float* wsoft, const float* o_re, const float* o_im,
                       float* outreal) {
  __shared__ float red[256];
  int j = blockIdx.x, t = threadIdx.x;
  const float* src = (j < D_OUT) ? o_re : o_im;
  int col = j & (D_OUT - 1);
  float s = 0.f;
  for (int n = t; n < N_CELLS; n += 256)
    s += wsoft[n] * src[(size_t)n * D_OUT + col];
  red[t] = s; __syncthreads();
  for (int k = 128; k > 0; k >>= 1) { if (t < k) red[t] += red[t + k]; __syncthreads(); }
  if (t == 0) outreal[j] = red[0];
}

// pred = out_real @ oh_w + oh_b   ([1x1024] x [1024x512])
__global__ void k_pred(const float* outreal, const float* oh_w, const float* oh_b,
                       float* pred) {
  int i = blockIdx.x * blockDim.x + threadIdx.x;
  if (i >= D_IN) return;
  float s = oh_b[i];
  for (int j = 0; j < 2 * D_OUT; ++j) s += outreal[j] * oh_w[(size_t)j * D_IN + i];
  pred[i] = s;
}

// ---------------- host launcher ----------------
extern "C" void kernel_launch(void* const* d_in, const int* in_sizes, int n_in,
                              void* d_out, int out_size, void* d_ws, size_t ws_size,
                              hipStream_t stream) {
  const float* x     = (const float*)d_in[0];
  const float* h_re  = (const float*)d_in[1];
  const float* h_im  = (const float*)d_in[2];
  const float* ea_wr = (const float*)d_in[3];
  const float* ea_br = (const float*)d_in[4];
  const float* ea_wi = (const float*)d_in[5];
  const float* ea_bi = (const float*)d_in[6];
  const float* eg_wr = (const float*)d_in[7];
  const float* eg_br = (const float*)d_in[8];
  const float* eg_wi = (const float*)d_in[9];
  const float* eg_bi = (const float*)d_in[10];
  const float* gz_w  = (const float*)d_in[11];
  const float* gz_b  = (const float*)d_in[12];
  const float* gr_w  = (const float*)d_in[13];
  const float* gr_b  = (const float*)d_in[14];
  const float* ghr_w = (const float*)d_in[15];
  const float* ghr_b = (const float*)d_in[16];
  const float* ghi_w = (const float*)d_in[17];
  const float* ghi_b = (const float*)d_in[18];
  const float* oh_w  = (const float*)d_in[19];
  const float* oh_b  = (const float*)d_in[20];
  const int*   step  = (const int*)d_in[21];

  // output layout: pred[512] | tmean[1] | nh_re[2048*1024] | nh_im[2048*1024]
  float* out   = (float*)d_out;
  float* pred  = out;
  float* tmean = out + D_IN;
  float* nh_re = out + D_IN + 1;
  float* nh_im = nh_re + (size_t)N_CELLS * D_HID;

  // workspace carve (256B aligned)
  char* p = (char*)d_ws;
  auto carve = [&](size_t bytes) -> void* {
    void* r = (void*)p;
    p += (bytes + 255) & ~(size_t)255;
    return r;
  };
  _Float16* hre16   = (_Float16*)carve((size_t)N_CELLS * D_HID * 2);
  _Float16* him16   = (_Float16*)carve((size_t)N_CELLS * D_HID * 2);
  _Float16* wrT     = (_Float16*)carve((size_t)D_OUT * D_HID * 2);
  _Float16* wiT     = (_Float16*)carve((size_t)D_OUT * D_HID * 2);
  _Float16* cm16    = (_Float16*)carve((size_t)N_CELLS * GIN_PAD * 2);
  _Float16* gzT     = (_Float16*)carve((size_t)D_HID * GIN_PAD * 2);
  _Float16* grT     = (_Float16*)carve((size_t)D_HID * GIN_PAD * 2);
  _Float16* combr16 = (_Float16*)carve((size_t)N_CELLS * GIN_PAD * 2);
  _Float16* combi16 = (_Float16*)carve((size_t)N_CELLS * GIN_PAD * 2);
  _Float16* ghrT    = (_Float16*)carve((size_t)D_HID * GIN_PAD * 2);
  _Float16* ghiT    = (_Float16*)carve((size_t)D_HID * GIN_PAD * 2);
  float* o_re    = (float*)carve((size_t)N_CELLS * D_OUT * 4);
  float* o_im    = (float*)carve((size_t)N_CELLS * D_OUT * 4);
  float* tension = (float*)carve((size_t)N_CELLS * 4);
  float* xa_re   = (float*)carve((size_t)D_OUT * 4);
  float* xa_im   = (float*)carve((size_t)D_OUT * 4);
  float* zbuf    = (float*)carve((size_t)N_CELLS * D_HID * 4);
  float* rbuf    = (float*)carve((size_t)N_CELLS * D_HID * 4);
  float* fmean   = (float*)carve((size_t)2 * NFAC * D_HID * 4);
  float* gmean   = (float*)carve((size_t)2 * D_HID * 4);
  float* wsoft   = (float*)carve((size_t)N_CELLS * 4);
  float* outreal = (float*)carve((size_t)2 * D_OUT * 4);

  const int B = 256;

  // prep / conversions
  k_conv_h<<<4096, B, 0, stream>>>(h_re, h_im, hre16, him16);
  k_conv_engine_w<<<2048, B, 0, stream>>>(ea_wr, eg_wr, ea_wi, eg_wi, wrT, wiT);
  k_xa<<<2, B, 0, stream>>>(x, ea_wr, eg_wr, ea_wi, eg_wi,
                            ea_br, eg_br, ea_bi, eg_bi, xa_re, xa_im);

  // engine: o_re, o_im  (2048 wave-tiles / 8 waves per block)
  k_engine_gemm<<<256, B, 0, stream>>>(hre16, him16, wrT, wiT, xa_re, xa_im, o_re, o_im);
  k_tension<<<N_CELLS, B, 0, stream>>>(o_re, o_im, tension);

  // gates
  k_build_cm<<<6272, B, 0, stream>>>(o_re, o_im, tension, h_re, h_im, cm16);
  k_conv_wT_pad<<<6272, B, 0, stream>>>(gz_w, gzT, GIN, D_HID, GIN_PAD);
  k_conv_wT_pad<<<6272, B, 0, stream>>>(gr_w, grT, GIN, D_HID, GIN_PAD);
  k_zr_gemm<<<512, B, 0, stream>>>(cm16, gzT, grT, gz_b, gr_b, zbuf, rbuf);

  // candidates -> nh (pre-sync) into d_out
  k_build_comb<<<6272, B, 0, stream>>>(o_re, o_im, tension, rbuf, h_re, h_im,
                                       combr16, combi16);
  k_conv_wT_pad<<<6272, B, 0, stream>>>(ghr_w, ghrT, GIN, D_HID, GIN_PAD);
  k_conv_wT_pad<<<6272, B, 0, stream>>>(ghi_w, ghiT, GIN, D_HID, GIN_PAD);
  k_cand_gemm<<<512, B, 0, stream>>>(combr16, combi16, ghrT, ghiT, ghr_b, ghi_b,
                                     zbuf, h_re, h_im, nh_re, nh_im);

  // faction sync + debate (in place on d_out)
  k_fmean<<<16, B, 0, stream>>>(nh_re, nh_im, fmean);
  k_gmean<<<8, B, 0, stream>>>(fmean, gmean);
  k_blend<<<8192, B, 0, stream>>>(nh_re, nh_im, fmean, gmean, step);

  // softmax-weighted combine + prediction head
  k_softmax<<<1, B, 0, stream>>>(tension, wsoft, tmean);
  k_wsum<<<2 * D_OUT, B, 0, stream>>>(wsoft, o_re, o_im, outreal);
  k_pred<<<2, B, 0, stream>>>(outreal, oh_w, oh_b, pred);
}
